// MoE_18382460027104
// MI455X (gfx1250) — compile-verified, matrix operand order checked by reading, not run on hardware
//
#include <hip/hip_runtime.h>
#include <cstdint>
#include <cstddef>

#define D_MODEL 1024
#define D_MLP   2048
#define NEXP    8
#define TM      32          // tokens per tile
#define NT      512         // threads per MoE workgroup (16 waves)

typedef __attribute__((ext_vector_type(16))) __bf16 v16bf;
typedef __attribute__((ext_vector_type(8)))  float  v8f;
typedef __attribute__((ext_vector_type(16))) unsigned short u16x16;
typedef __attribute__((ext_vector_type(8)))  unsigned short u16x8;

__device__ __forceinline__ unsigned short f2bf(float f) {
  unsigned int u = __float_as_uint(f);
  u += 0x7FFFu + ((u >> 16) & 1u);   // round-to-nearest-even fp32 -> bf16
  return (unsigned short)(u >> 16);
}

__device__ __forceinline__ unsigned int pack2(float a, float b) {
  return (unsigned int)f2bf(a) | ((unsigned int)f2bf(b) << 16);
}

__device__ __forceinline__ unsigned long long pack4(float a, float b, float c, float d) {
  return (unsigned long long)pack2(a, b) | ((unsigned long long)pack2(c, d) << 32);
}

// Assemble a 16x32 (or 32x16) bf16 WMMA fragment from an LDS row.
// Lane layout (16-bit A/B): lane = g*16 + m; per lane K = {g*8..g*8+7, 16+g*8..16+g*8+7}.
__device__ __forceinline__ v16bf load_frag(const unsigned short* row, int g) {
  u16x8 lo = *(const u16x8*)(row + g * 8);
  u16x8 hi = *(const u16x8*)(row + 16 + g * 8);
  u16x16 u;
#pragma unroll
  for (int i = 0; i < 8; ++i) { u[i] = lo[i]; u[i + 8] = hi[i]; }
  return __builtin_bit_cast(v16bf, u);
}

// ---------------- Router: logits, top-2, expert counts ----------------
__global__ __launch_bounds__(256) void router_kernel(
    const float* __restrict__ x, const float* __restrict__ Wr,
    int* __restrict__ topIdx, float* __restrict__ topW,
    int* __restrict__ counts, float* __restrict__ logits, int T)
{
  const int wave = threadIdx.x >> 5, lane = threadIdx.x & 31;
  const int t = blockIdx.x * 8 + wave;
  if (t >= T) return;
  const float* h = x + (size_t)t * D_MODEL;
  float acc[NEXP];
#pragma unroll
  for (int e = 0; e < NEXP; ++e) acc[e] = 0.f;
  for (int d = lane; d < D_MODEL; d += 32) {
    float hv = h[d];
    const float* wr = Wr + (size_t)d * NEXP;
#pragma unroll
    for (int e = 0; e < NEXP; ++e) acc[e] = fmaf(hv, wr[e], acc[e]);
  }
#pragma unroll
  for (int e = 0; e < NEXP; ++e) {
    float v = acc[e];
#pragma unroll
    for (int off = 16; off > 0; off >>= 1) v += __shfl_xor(v, off, 32);
    acc[e] = v;
  }
  if (lane == 0) {
#pragma unroll
    for (int e = 0; e < NEXP; ++e) logits[(size_t)t * NEXP + e] = acc[e];
    int i0 = 0;
#pragma unroll
    for (int e = 1; e < NEXP; ++e) if (acc[e] > acc[i0]) i0 = e;
    int i1 = (i0 == 0) ? 1 : 0;
#pragma unroll
    for (int e = 0; e < NEXP; ++e) if (e != i0 && acc[e] > acc[i1]) i1 = e;
    // softmax + top2 + renorm  ==  w0 = 1/(1+exp(l1-l0))
    float w0 = 1.f / (1.f + __expf(acc[i1] - acc[i0]));
    topIdx[t * 2 + 0] = i0; topW[t * 2 + 0] = w0;
    topIdx[t * 2 + 1] = i1; topW[t * 2 + 1] = 1.f - w0;
    atomicAdd(&counts[i0], 1);
    atomicAdd(&counts[i1], 1);
  }
}

// ---------------- Tile-offset scan (8 experts, trivial) ----------------
__global__ void scan_kernel(const int* __restrict__ counts, int* __restrict__ tileOff) {
  if (threadIdx.x == 0 && blockIdx.x == 0) {
    int acc = 0;
#pragma unroll
    for (int e = 0; e < NEXP; ++e) { tileOff[e] = acc; acc += (counts[e] + TM - 1) / TM; }
    tileOff[NEXP] = acc;
  }
}

// ---------------- Scatter tokens into per-expert compacted lists ----------------
__global__ __launch_bounds__(256) void scatter_kernel(
    const int* __restrict__ topIdx, const float* __restrict__ topW,
    const int* __restrict__ tileOff, int* __restrict__ cursor,
    int* __restrict__ sortedIdx, float* __restrict__ sortedW, int T)
{
  int t = blockIdx.x * 256 + threadIdx.x;
  if (t >= T) return;
#pragma unroll
  for (int k = 0; k < 2; ++k) {
    int e = topIdx[t * 2 + k];
    int p = atomicAdd(&cursor[e], 1);
    int dst = tileOff[e] * TM + p;
    sortedIdx[dst] = t;
    sortedW[dst]   = topW[t * 2 + k];
  }
}

// ---------------- Fused expert MLP over one 32-token tile ----------------
__global__ __launch_bounds__(NT) void moe_kernel(
    const float* __restrict__ x,
    const float* __restrict__ Wg, const float* __restrict__ Wi, const float* __restrict__ Wo,
    const int* __restrict__ sortedIdx, const float* __restrict__ sortedW,
    const int* __restrict__ tileOff, float* __restrict__ out)
{
  __shared__ int   s_tok[TM];
  __shared__ float s_w[TM];
  __shared__ __align__(16) unsigned short s_act[TM * 136];    // [32][128] padded to 136
  __shared__ __align__(16) unsigned short s_stage[512 * 40];  // 40 KB, aliased per phase

  const int tid  = threadIdx.x;
  const int wave = tid >> 5, lane = tid & 31;
  const int b    = blockIdx.x;
  const int g    = lane >> 4, r = lane & 15;
  const int wq   = wave & 7;          // n-group (8 groups)
  const int wm   = wave >> 3;         // m-tile (0 or 1)

  if (tid < TM) {
    s_tok[tid] = sortedIdx[b * TM + tid];
    s_w[tid]   = sortedW[b * TM + tid];
  }
  __syncthreads();

  float tot = 0.f;
#pragma unroll
  for (int i = 0; i < TM; ++i) tot += fabsf(s_w[i]);
  if (tot == 0.f) return;  // uniform: all-padding tile

  // expert owning this tile: max e with tileOff[e] <= b
  int e = 0;
#pragma unroll
  for (int q = 1; q < NEXP; ++q) if (tileOff[q] <= b) e = q;

  const float* Wg_e = Wg + (size_t)e * D_MODEL * D_MLP;
  const float* Wi_e = Wi + (size_t)e * D_MODEL * D_MLP;
  const float* Wo_e = Wo + (size_t)e * D_MLP * D_MODEL;

  unsigned short (*s_a)[40]  = (unsigned short(*)[40])s_stage;                     // 32 rows
  unsigned short (*s_bg)[40] = (unsigned short(*)[40])(s_stage + TM * 40);         // 128 rows
  unsigned short (*s_bi)[40] = (unsigned short(*)[40])(s_stage + (TM + 128) * 40); // 128 rows
  unsigned short (*s_bo)[40] = (unsigned short(*)[40])s_stage;                     // 512 rows (phase 2)

  // A-staging: thread owns (row am = tid>>4, col pair ak = (tid&15)*2); hoist row ptr.
  const int am = tid >> 4;            // 0..31
  const int ak = (tid & 15) * 2;
  const float* aRow = x + (size_t)s_tok[am] * D_MODEL + ak;
  unsigned int* aDst = (unsigned int*)&s_a[am][ak];

  v8f accO[8];
#pragma unroll
  for (int i = 0; i < 8; ++i) {
#pragma unroll
    for (int j = 0; j < 8; ++j) accO[i][j] = 0.f;
  }

#pragma unroll 1
  for (int f0 = 0; f0 < D_MLP; f0 += 128) {
    v8f accG, accU;
#pragma unroll
    for (int j = 0; j < 8; ++j) { accG[j] = 0.f; accU[j] = 0.f; }

    // ---- Phase 1: G = h*Wg, U = h*Wi for 32 tokens x 128 F-cols ----
#pragma unroll 1
    for (int k0 = 0; k0 < D_MODEL; k0 += 32) {
      // A tile: one float2 load + one packed b32 LDS store per thread
      {
        float2 v = *(const float2*)(aRow + k0);
        *aDst = pack2(v.x, v.y);
      }
      // Wg/Wi 32x128 tiles -> transposed [n][k]: 2 units/thread,
      // each unit = 4 coalesced b32 loads (fixed n, 4 consecutive k) + 1 b64 LDS store
#pragma unroll
      for (int it = 0; it < 2; ++it) {
        int u  = tid + it * NT;        // 0..1023
        int kq = u >> 7;               // 0..7 -> k = 4*kq
        int n  = u & 127;
        const float* pg = Wg_e + (size_t)(k0 + kq * 4) * D_MLP + f0 + n;
        const float* pi = Wi_e + (size_t)(k0 + kq * 4) * D_MLP + f0 + n;
        float g0 = pg[0], g1 = pg[D_MLP], g2 = pg[2 * D_MLP], g3 = pg[3 * D_MLP];
        float u0 = pi[0], u1 = pi[D_MLP], u2 = pi[2 * D_MLP], u3 = pi[3 * D_MLP];
        *(unsigned long long*)&s_bg[n][kq * 4] = pack4(g0, g1, g2, g3);
        *(unsigned long long*)&s_bi[n][kq * 4] = pack4(u0, u1, u2, u3);
      }
      if (k0 + 32 < D_MODEL)  // speculative prefetch of next weight slab
        __builtin_prefetch(Wg_e + (size_t)(k0 + 32 + (tid >> 4)) * D_MLP + f0 + ((tid & 15) << 3), 0, 1);
      __syncthreads();

      v16bf af  = load_frag(s_a[wm * 16 + r], g);
      v16bf bgf = load_frag(s_bg[wq * 16 + r], g);
      v16bf bif = load_frag(s_bi[wq * 16 + r], g);
      accG = __builtin_amdgcn_wmma_f32_16x16x32_bf16(false, af, false, bgf, (short)0, accG, false, false);
      accU = __builtin_amdgcn_wmma_f32_16x16x32_bf16(false, af, false, bif, (short)0, accU, false, false);
      __syncthreads();
    }

    // ---- act = silu(G) * U * route_weight -> LDS bf16 ----
#pragma unroll
    for (int r2 = 0; r2 < 8; ++r2) {
      int m = wm * 16 + r2 + g * 8;        // C/D layout: lanes 0-15 M=r2, lanes 16-31 M=8+r2
      float gv = accG[r2], uv = accU[r2];
      float a = (gv / (1.f + __expf(-gv))) * uv * s_w[m];
      s_act[m * 136 + wq * 16 + r] = f2bf(a);
    }
    __syncthreads();

    // ---- Phase 2: OUT += ACT(32x128) * Wo(128x1024) ----
#pragma unroll 1
    for (int kc = 0; kc < 4; ++kc) {
      v16bf aa = load_frag(&s_act[(wm * 16 + r) * 136 + kc * 32], g);
#pragma unroll
      for (int h = 0; h < 2; ++h) {
        // stage Wo 32x512 slab transposed to [n][k]: 8 units/thread,
        // each unit = 4 coalesced b32 loads + 1 b64 LDS store
#pragma unroll 4
        for (int it = 0; it < 8; ++it) {
          int u  = tid + it * NT;      // 0..4095
          int kq = u >> 9;             // 0..7 -> k = 4*kq
          int n  = u & 511;
          const float* po = Wo_e + (size_t)(f0 + kc * 32 + kq * 4) * D_MODEL + h * 512 + n;
          *(unsigned long long*)&s_bo[n][kq * 4] =
              pack4(po[0], po[D_MODEL], po[2 * D_MODEL], po[3 * D_MODEL]);
        }
        __syncthreads();
#pragma unroll
        for (int j = 0; j < 4; ++j) {
          v16bf bb = load_frag(s_bo[(wq * 4 + j) * 16 + r], g);
          accO[h * 4 + j] = __builtin_amdgcn_wmma_f32_16x16x32_bf16(
              false, aa, false, bb, (short)0, accO[h * 4 + j], false, false);
        }
        __syncthreads();
      }
    }
  }

  // ---- scatter-add: each token appears in exactly 2 expert groups ----
#pragma unroll
  for (int h = 0; h < 2; ++h) {
#pragma unroll
    for (int j = 0; j < 4; ++j) {
      int col = h * 512 + (wq * 4 + j) * 16 + r;
#pragma unroll
      for (int r2 = 0; r2 < 8; ++r2) {
        int m = wm * 16 + r2 + g * 8;
        if (s_w[m] != 0.f)
          unsafeAtomicAdd(&out[(size_t)s_tok[m] * D_MODEL + col], accO[h * 4 + j][r2]);
      }
    }
  }
}

// ---------------- Host launcher ----------------
extern "C" void kernel_launch(void* const* d_in, const int* in_sizes, int n_in,
                              void* d_out, int out_size, void* d_ws, size_t ws_size,
                              hipStream_t stream) {
  const float* x  = (const float*)d_in[0];
  const float* Wr = (const float*)d_in[1];
  const float* Wg = (const float*)d_in[2];
  const float* Wi = (const float*)d_in[3];
  const float* Wo = (const float*)d_in[4];
  float* out = (float*)d_out;

  const int T = in_sizes[0] / D_MODEL;              // 4096 tokens
  const int tiles = (T * 2) / TM + NEXP;            // padded tile upper bound (264)

  // workspace layout (bytes)
  char* ws = (char*)d_ws;
  int*   counts    = (int*)(ws + 0);                //  8 ints
  int*   cursor    = (int*)(ws + 32);               //  8 ints
  int*   tileOff   = (int*)(ws + 64);               //  9 ints (pad to 16)
  int*   sortedIdx = (int*)(ws + 128);              //  tiles*TM ints
  float* sortedW   = (float*)(ws + 128 + (size_t)tiles * TM * 4);
  int*   topIdx    = (int*)(ws + 128 + (size_t)tiles * TM * 8);
  float* topW      = (float*)(ws + 128 + (size_t)tiles * TM * 8 + (size_t)T * 8);
  size_t zeroBytes = 128 + (size_t)tiles * TM * 8;  // counts..sortedW

  float* logits = out + (size_t)T * D_MODEL;        // second tuple output

  hipMemsetAsync(d_ws, 0, zeroBytes, stream);
  hipMemsetAsync(out, 0, (size_t)T * D_MODEL * sizeof(float), stream);

  router_kernel<<<dim3((T + 7) / 8), dim3(256), 0, stream>>>(x, Wr, topIdx, topW, counts, logits, T);
  scan_kernel<<<dim3(1), dim3(32), 0, stream>>>(counts, tileOff);
  scatter_kernel<<<dim3((T + 255) / 256), dim3(256), 0, stream>>>(topIdx, topW, tileOff, cursor,
                                                                  sortedIdx, sortedW, T);
  moe_kernel<<<dim3(tiles), dim3(NT), 0, stream>>>(x, Wg, Wi, Wo, sortedIdx, sortedW, tileOff, out);
}
